// MultiHeadAttentionClassical_65481071409218
// MI455X (gfx1250) — matrix operand order repaired and checked by reading, not compiled, not run on hardware
//
#include <hip/hip_runtime.h>
#include <hip/hip_bf16.h>

typedef __attribute__((ext_vector_type(16))) _Float16 v16h;
typedef __attribute__((ext_vector_type(8)))  float    v8f;

#define WMMA_F16(A, B, C) \
    __builtin_amdgcn_wmma_f32_16x16x32_f16(false, (A), false, (B), (short)0, (C), false, false)

// ---------------------------------------------------------------------------
// Problem constants (match reference)
// ---------------------------------------------------------------------------
constexpr int S_LEN = 2048;
constexpr int EMB   = 1024;
constexpr int HEADS = 16;
constexpr int DK    = 64;     // EMB / HEADS

// ---------------------------------------------------------------------------
// fp32 -> fp16 conversion (grid-stride)
// ---------------------------------------------------------------------------
__global__ void cvt_f32_f16(const float* __restrict__ src, _Float16* __restrict__ dst, int n) {
    for (int i = blockIdx.x * blockDim.x + threadIdx.x; i < n; i += gridDim.x * blockDim.x)
        dst[i] = (_Float16)src[i];
}

// ---------------------------------------------------------------------------
// C[M,N] = A[M,K] * B[N,K]^T   (A,B f16 row-major; contraction contiguous in both)
// Block: 256 threads = 8 waves arranged 2(M) x 4(N); wave tile = 64x64
// (4x4 accumulator tiles -> 16 WMMAs per 8 fragment loads, ~32 FLOP/B).
// Block tile = 128 x 256.  Operand reuse is served by the 192MB L2 ->
// direct global fragment loads, no LDS staging.
// ---------------------------------------------------------------------------
template <bool OUT_F32>
__global__ void __launch_bounds__(256)
gemm_nt_f16(const _Float16* __restrict__ A, const _Float16* __restrict__ Bm,
            void* __restrict__ Cout, int M, int N, int K) {
    const int lane = threadIdx.x & 31;
    const int w    = threadIdx.x >> 5;
    const int wm   = w & 1;        // 0..1
    const int wn   = w >> 1;       // 0..3
    const int mb   = blockIdx.x * 128 + wm * 64;
    const int nb   = blockIdx.y * 256 + wn * 64;
    const int lm   = lane & 15;
    const int lh   = lane >> 4;

    v8f acc[4][4] = {};

    for (int k0 = 0; k0 < K; k0 += 32) {
        v16h a[4], b[4];
#pragma unroll
        for (int i = 0; i < 4; ++i)
            a[i] = *(const v16h*)(A + (size_t)(mb + i * 16 + lm) * K + k0 + lh * 16);
#pragma unroll
        for (int j = 0; j < 4; ++j)
            b[j] = *(const v16h*)(Bm + (size_t)(nb + j * 16 + lm) * K + k0 + lh * 16);
#pragma unroll
        for (int i = 0; i < 4; ++i)
#pragma unroll
            for (int j = 0; j < 4; ++j)
                acc[i][j] = WMMA_F16(a[i], b[j], acc[i][j]);
    }

#pragma unroll
    for (int i = 0; i < 4; ++i)
#pragma unroll
        for (int j = 0; j < 4; ++j)
#pragma unroll
            for (int r = 0; r < 8; ++r) {
                const int row = mb + i * 16 + r + 8 * lh;   // C layout: VGPR r -> M=r / r+8
                const int col = nb + j * 16 + lm;           // lanes 0-15 / 16-31 -> N
                if (OUT_F32)
                    ((float*)Cout)[(size_t)row * N + col] = acc[i][j][r];
                else
                    ((_Float16*)Cout)[(size_t)row * N + col] = (_Float16)acc[i][j][r];
            }
}

// ---------------------------------------------------------------------------
// Flash attention: O = softmax(Q K^T / sqrt(dk)) V, per (batch, head).
// One wave owns a 16-row query tile; block = 8 waves = 128 query rows.
// Streams keys/values in chunks of 64 with online softmax (reduction loop
// and accumulator rescale amortized over 64 keys -> 16 WMMAs per reduction).
// Q/K/V/O stored as [b*S + s][h*DK + d] f16 (row stride EMB).
// ---------------------------------------------------------------------------
__global__ void __launch_bounds__(256)
flash_attn_f16(const _Float16* __restrict__ Q, const _Float16* __restrict__ K,
               const _Float16* __restrict__ V, _Float16* __restrict__ O) {
    __shared__ _Float16 plds[8][16 * 64];   // per-wave P-tile transpose staging (16KB)

    const int lane = threadIdx.x & 31;
    const int w    = threadIdx.x >> 5;
    const int lm   = lane & 15;
    const int lh   = lane >> 4;

    const int b  = blockIdx.z;
    const int h  = blockIdx.y;
    const int q0 = blockIdx.x * 128 + w * 16;

    const _Float16* Qb = Q + (size_t)b * S_LEN * EMB + h * DK;
    const _Float16* Kb = K + (size_t)b * S_LEN * EMB + h * DK;
    const _Float16* Vb = V + (size_t)b * S_LEN * EMB + h * DK;
    _Float16*       Ob = O + (size_t)b * S_LEN * EMB + h * DK;

    // Q fragments (16 rows x 64 dk) = two 16x32 A-fragments, kept in registers.
    const v16h qa0 = *(const v16h*)(Qb + (size_t)(q0 + lm) * EMB + lh * 16);
    const v16h qa1 = *(const v16h*)(Qb + (size_t)(q0 + lm) * EMB + 32 + lh * 16);

    float mrow[8], lrow[8];
    v8f   o[4] = {};
#pragma unroll
    for (int r = 0; r < 8; ++r) { mrow[r] = -1e30f; lrow[r] = 0.0f; }

    const float scale = 0.125f;   // 1/sqrt(64)

    for (int j = 0; j < S_LEN; j += 64) {
        // ---- scores: 4 tiles of 16 keys, dk=64 contraction = 2 WMMAs each ----
        v8f s[4];
#pragma unroll
        for (int c = 0; c < 4; ++c) {
            const _Float16* kr = Kb + (size_t)(j + c * 16 + lm) * EMB + lh * 16;
            const v16h kb0 = *(const v16h*)(kr);
            const v16h kb1 = *(const v16h*)(kr + 32);
            v8f sc = {};
            sc = WMMA_F16(qa0, kb0, sc);
            sc = WMMA_F16(qa1, kb1, sc);
            s[c] = sc;
        }

        // ---- online softmax over 64 keys (one reduction pass) ----
        // Row M = r + 8*lh lives across 16 lanes of acc element r.
#pragma unroll
        for (int r = 0; r < 8; ++r) {
            float cmax = fmaxf(fmaxf(s[0][r], s[1][r]), fmaxf(s[2][r], s[3][r]));
#pragma unroll
            for (int m = 1; m < 16; m <<= 1)
                cmax = fmaxf(cmax, __shfl_xor(cmax, m, 16));
            const float mnew = fmaxf(mrow[r], cmax);
            const float f    = __expf(scale * (mrow[r] - mnew));
            mrow[r] = mnew;

            float p[4], psum = 0.0f;
#pragma unroll
            for (int c = 0; c < 4; ++c) {
                p[c] = __expf(scale * (s[c][r] - mnew));
                psum += p[c];
            }
#pragma unroll
            for (int m = 1; m < 16; m <<= 1)
                psum += __shfl_xor(psum, m, 16);
            lrow[r] = lrow[r] * f + psum;

            o[0][r] *= f; o[1][r] *= f; o[2][r] *= f; o[3][r] *= f;

            // Stage P (C-layout) into LDS for re-read in A-fragment layout.
            const int prow = r + 8 * lh;
#pragma unroll
            for (int c = 0; c < 4; ++c)
                plds[w][prow * 64 + c * 16 + lm] = (_Float16)p[c];
        }

        // Re-load P(16x64) as two A-fragments (compiler inserts s_wait_dscnt).
        const v16h pa0 = *(const v16h*)(&plds[w][lm * 64 + lh * 16]);
        const v16h pa1 = *(const v16h*)(&plds[w][lm * 64 + 32 + lh * 16]);

        // ---- PV: V chunk 64 keys x 64 dims -> 2 key-halves x 4 col slices ----
#pragma unroll
        for (int t = 0; t < 4; ++t) {
            v16h vb0, vb1;
#pragma unroll
            for (int e = 0; e < 16; ++e) {
                vb0[e] = Vb[(size_t)(j + lh * 16 + e) * EMB + t * 16 + lm];
                vb1[e] = Vb[(size_t)(j + 32 + lh * 16 + e) * EMB + t * 16 + lm];
            }
            o[t] = WMMA_F16(pa0, vb0, o[t]);
            o[t] = WMMA_F16(pa1, vb1, o[t]);
        }
    }

    // Normalize and store (f16, same [s][h*DK+d] layout).
#pragma unroll
    for (int r = 0; r < 8; ++r) {
        const float inv = 1.0f / lrow[r];
        const int row   = q0 + r + 8 * lh;
#pragma unroll
        for (int t = 0; t < 4; ++t)
            Ob[(size_t)row * EMB + t * 16 + lm] = (_Float16)(o[t][r] * inv);
    }
}

// ---------------------------------------------------------------------------
// Launch
// ---------------------------------------------------------------------------
extern "C" void kernel_launch(void* const* d_in, const int* in_sizes, int n_in,
                              void* d_out, int out_size, void* d_ws, size_t ws_size,
                              hipStream_t stream) {
    const float* x  = (const float*)d_in[0];
    const float* Wq = (const float*)d_in[1];
    const float* Wk = (const float*)d_in[2];
    const float* Wv = (const float*)d_in[3];
    const float* Wo = (const float*)d_in[4];

    const int B = in_sizes[0] / (S_LEN * EMB);   // = 2
    const int M = B * S_LEN;                     // 4096 rows for all projections
    const int NW = EMB * EMB;                    // weight elements

    // Workspace layout (f16 tensors); everything fits in L2 (192MB).
    char* ws = (char*)d_ws;
    _Float16* xh  = (_Float16*)(ws);                                   //  8 MB
    _Float16* Wqh = (_Float16*)(ws + (size_t)M * EMB * 2);             //  2 MB
    _Float16* Wkh = Wqh + NW;
    _Float16* Wvh = Wkh + NW;
    _Float16* Woh = Wvh + NW;
    _Float16* Qh  = Woh + NW;                                          //  8 MB
    _Float16* Kh  = Qh + (size_t)M * EMB;
    _Float16* Vh  = Kh + (size_t)M * EMB;
    _Float16* Ah  = Vh + (size_t)M * EMB;                              //  attn out

    // 1) convert inputs to f16
    cvt_f32_f16<<<2048, 256, 0, stream>>>(x,  xh,  M * EMB);
    cvt_f32_f16<<<1024, 256, 0, stream>>>(Wq, Wqh, NW);
    cvt_f32_f16<<<1024, 256, 0, stream>>>(Wk, Wkh, NW);
    cvt_f32_f16<<<1024, 256, 0, stream>>>(Wv, Wvh, NW);
    cvt_f32_f16<<<1024, 256, 0, stream>>>(Wo, Woh, NW);

    // 2) projections: Q/K/V = x @ W^T  (f16 out)
    dim3 ggrid(M / 128, EMB / 256);
    gemm_nt_f16<false><<<ggrid, 256, 0, stream>>>(xh, Wqh, Qh, M, EMB, EMB);
    gemm_nt_f16<false><<<ggrid, 256, 0, stream>>>(xh, Wkh, Kh, M, EMB, EMB);
    gemm_nt_f16<false><<<ggrid, 256, 0, stream>>>(xh, Wvh, Vh, M, EMB, EMB);

    // 3) flash attention per (q-tile, head, batch)
    dim3 agrid(S_LEN / 128, HEADS, B);
    flash_attn_f16<<<agrid, 256, 0, stream>>>(Qh, Kh, Vh, Ah);

    // 4) output projection: out = attn @ Wo^T  (f32 out)
    gemm_nt_f16<true><<<ggrid, 256, 0, stream>>>(Ah, Woh, d_out, M, EMB, EMB);
}
